// GumbelPromptPool_11768210391457
// MI455X (gfx1250) — matrix operand order repaired and checked by reading, not compiled, not run on hardware
//
#include <hip/hip_runtime.h>
#include <hip/hip_bf16.h>

typedef __attribute__((ext_vector_type(16))) _Float16 v16h;
typedef __attribute__((ext_vector_type(8)))  float    v8f;

#define BB   256
#define SS   196
#define DD   1024
#define PP   512
#define LL   8
#define TOPK 4

// ---------------------------------------------------------------------------
// Kernel 1: query = mean over S of x_embed[b,:,:], l2-normalized, stored f16.
// One block per b; 256 threads, each owns 4 consecutive d via float4 loads.
// Coalesced 4 KB per (b,s) row read; this is the bandwidth-dominant pass.
// ---------------------------------------------------------------------------
__global__ void qmean_norm_kernel(const float* __restrict__ x,
                                  _Float16* __restrict__ q16) {
    const int b = blockIdx.x;
    const int t = threadIdx.x;
    const float4* row = (const float4*)(x + (size_t)b * SS * DD);
    float4 acc = {0.f, 0.f, 0.f, 0.f};
    for (int s = 0; s < SS; ++s) {
        float4 v = row[s * (DD / 4) + t];
        acc.x += v.x; acc.y += v.y; acc.z += v.z; acc.w += v.w;
    }
    const float inv = 1.0f / (float)SS;
    acc.x *= inv; acc.y *= inv; acc.z *= inv; acc.w *= inv;

    __shared__ float red[256];
    red[t] = acc.x * acc.x + acc.y * acc.y + acc.z * acc.z + acc.w * acc.w;
    __syncthreads();
    for (int st = 128; st > 0; st >>= 1) {
        if (t < st) red[t] += red[t + st];
        __syncthreads();
    }
    const float scale = rsqrtf(fmaxf(red[0], 1e-12f));
    _Float16* o = q16 + (size_t)b * DD + t * 4;
    o[0] = (_Float16)(acc.x * scale);
    o[1] = (_Float16)(acc.y * scale);
    o[2] = (_Float16)(acc.z * scale);
    o[3] = (_Float16)(acc.w * scale);
}

// ---------------------------------------------------------------------------
// Kernel 2: key_n = l2_normalize(prompt_key, axis=1), stored f16. One block
// per pool row (512 blocks); tiny (2 MB read).
// ---------------------------------------------------------------------------
__global__ void keynorm_kernel(const float* __restrict__ pkey,
                               _Float16* __restrict__ k16) {
    const int p = blockIdx.x;
    const int t = threadIdx.x;
    const float4* row = (const float4*)(pkey + (size_t)p * DD);
    float4 v = row[t];

    __shared__ float red[256];
    red[t] = v.x * v.x + v.y * v.y + v.z * v.z + v.w * v.w;
    __syncthreads();
    for (int st = 128; st > 0; st >>= 1) {
        if (t < st) red[t] += red[t + st];
        __syncthreads();
    }
    const float scale = rsqrtf(fmaxf(red[0], 1e-12f));
    _Float16* o = k16 + (size_t)p * DD + t * 4;
    o[0] = (_Float16)(v.x * scale);
    o[1] = (_Float16)(v.y * scale);
    o[2] = (_Float16)(v.z * scale);
    o[3] = (_Float16)(v.w * scale);
}

// ---------------------------------------------------------------------------
// Kernel 3: sim[256,512] = q_n @ key_n^T via V_WMMA_F32_16X16X32_F16.
// One wave32 per 16x16 output tile; grid = (512/16, 256/16) = (32, 16).
// K = 1024 swept in steps of 32 -> 32 v_wmma per wave, fp32 accumulation.
// EXEC is all-1s (blockDim = 32, no divergence), as WMMA requires.
// ---------------------------------------------------------------------------
__global__ void sim_wmma_kernel(const _Float16* __restrict__ q16,
                                const _Float16* __restrict__ k16,
                                float* __restrict__ sim) {
    const int lane  = threadIdx.x;        // 0..31
    const int ptile = blockIdx.x;         // 0..31 (N tiles)
    const int btile = blockIdx.y;         // 0..15 (M tiles)
    const int hi    = lane >> 4;          // 0 or 1 (lane half)
    const int l15   = lane & 15;

    const _Float16* qrow = q16 + (size_t)(btile * 16 + l15) * DD; // A row (M)
    const _Float16* krow = k16 + (size_t)(ptile * 16 + l15) * DD; // B col (N) = key row
    const int kaBase = hi * 8;   // A: lanes 16-31 start at K=8 within tile
    const int kbBase = hi * 8;   // B: lanes 16-31 start at K=8 within tile

    v8f acc = {};
    for (int k0 = 0; k0 < DD; k0 += 32) {
        v16h a, bm;
#pragma unroll
        for (int v = 0; v < 8; ++v) {
            const int kA = k0 + kaBase + 2 * (v & 3) + ((v >> 2) << 4);
            a[2 * v]     = qrow[kA];
            a[2 * v + 1] = qrow[kA + 1];
            const int kB = k0 + kbBase + 2 * (v & 3) + ((v >> 2) << 4);
            bm[2 * v]     = krow[kB];
            bm[2 * v + 1] = krow[kB + 1];
        }
        acc = __builtin_amdgcn_wmma_f32_16x16x32_f16(
            /*neg_a=*/false, a, /*neg_b=*/false, bm,
            /*c_mod=*/(short)0, acc, /*reuse_a=*/false, /*reuse_b=*/false);
    }
#pragma unroll
    for (int r = 0; r < 8; ++r) {
        const int m = btile * 16 + r + hi * 8;
        const int n = ptile * 16 + l15;
        sim[(size_t)m * PP + n] = acc[r];
    }
}

// ---------------------------------------------------------------------------
// Kernel 4: fused Gumbel top-k selection + prompt gather. One block per b.
// Forward value of the straight-through estimator is the hard one-hot, so
// each round is argmax_p(cur[p] + g_i[p]); winner gets cur -= 1000.
// Ties break to the lowest index (matches jnp.argmax).
// Then copy prompt[sel[i]] (32 KB contiguous) to out[b, i*8:(i+1)*8, :].
// ---------------------------------------------------------------------------
__global__ void select_gather_kernel(const float* __restrict__ sim,
                                     const float* __restrict__ gum,
                                     const float* __restrict__ prompt,
                                     float* __restrict__ out) {
    const int b = blockIdx.x;
    const int t = threadIdx.x;
    __shared__ float sv[256];
    __shared__ int   si[256];

    float cur0 = sim[(size_t)b * PP + t];
    float cur1 = sim[(size_t)b * PP + t + 256];
    int sel[TOPK];

    for (int i = 0; i < TOPK; ++i) {
        const float* gu = gum + ((size_t)i * BB + b) * PP;
        const float g0 = -logf(-logf(gu[t]       + 1e-10f) + 1e-10f);
        const float g1 = -logf(-logf(gu[t + 256] + 1e-10f) + 1e-10f);
        const float l0 = cur0 + g0;
        const float l1 = cur1 + g1;
        float bv; int bi;
        if (l1 > l0) { bv = l1; bi = t + 256; } else { bv = l0; bi = t; }
        sv[t] = bv; si[t] = bi;
        __syncthreads();
        for (int st = 128; st > 0; st >>= 1) {
            if (t < st) {
                const float ov = sv[t + st];
                const int   oi = si[t + st];
                if (ov > sv[t] || (ov == sv[t] && oi < si[t])) {
                    sv[t] = ov; si[t] = oi;
                }
            }
            __syncthreads();
        }
        const int s = si[0];
        __syncthreads();   // allow sv/si reuse next round
        sel[i] = s;
        if (s == t)       cur0 -= 1000.0f;
        if (s == t + 256) cur1 -= 1000.0f;
    }

    // Gather: 4 x (8*1024 floats) = 32 KB per (b,i), float4 copies.
    float4* dst = (float4*)(out + (size_t)b * TOPK * LL * DD);
    for (int i = 0; i < TOPK; ++i) {
        const float4* src = (const float4*)(prompt + (size_t)sel[i] * LL * DD);
        for (int tt = t; tt < (LL * DD / 4); tt += 256)
            dst[i * (LL * DD / 4) + tt] = src[tt];
    }
}

// ---------------------------------------------------------------------------
extern "C" void kernel_launch(void* const* d_in, const int* in_sizes, int n_in,
                              void* d_out, int out_size, void* d_ws, size_t ws_size,
                              hipStream_t stream) {
    (void)in_sizes; (void)n_in; (void)out_size; (void)ws_size;
    const float* x_embed = (const float*)d_in[0]; // [256,196,1024]
    const float* prompt  = (const float*)d_in[1]; // [512,8,1024]
    const float* pkey    = (const float*)d_in[2]; // [512,1024]
    const float* gum     = (const float*)d_in[3]; // [4,256,512]
    float* out = (float*)d_out;                   // [256,32,1024]

    char* ws = (char*)d_ws;
    _Float16* q16 = (_Float16*)ws;                                   // 512 KB
    _Float16* k16 = (_Float16*)(ws + (size_t)BB * DD * 2);           // 1 MB
    float*    sim = (float*)(ws + (size_t)BB * DD * 2
                                + (size_t)PP * DD * 2);              // 512 KB

    qmean_norm_kernel<<<BB, 256, 0, stream>>>(x_embed, q16);
    keynorm_kernel<<<PP, 256, 0, stream>>>(pkey, k16);
    sim_wmma_kernel<<<dim3(PP / 16, BB / 16), 32, 0, stream>>>(q16, k16, sim);
    select_gather_kernel<<<BB, 256, 0, stream>>>(sim, gum, prompt, out);
}
